// GCN_17600775979728
// MI455X (gfx1250) — matrix operand order repaired and checked by reading, not compiled, not run on hardware
//
#include <hip/hip_runtime.h>
#include <hip/hip_bf16.h>
#include <math.h>

// ---------------------------------------------------------------------------
// GraphConv stack for MI455X (gfx1250): bf16 WMMA GEMMs + L2-resident f32
// atomic edge aggregation + deterministic pooling + f32 MLP tail.
// ---------------------------------------------------------------------------

#define N_NODES 32768
#define N_EDGES 524288
#define F_IN    32
#define HID     512
#define N_CLS   10
#define N_GRAPH 64

typedef __attribute__((ext_vector_type(16))) __bf16 v16bf;
typedef __attribute__((ext_vector_type(8)))  float  v8f;

union FragBF { v16bf v; uint4 u[2]; };

// ---------------------------------------------------------------------------
// Elementwise helpers
// ---------------------------------------------------------------------------
__global__ void zero_f32(float* p, int n) {
    int i = blockIdx.x * blockDim.x + threadIdx.x;
    if (i < n) p[i] = 0.0f;
}

__global__ void zero_i32(int* p, int n) {
    int i = blockIdx.x * blockDim.x + threadIdx.x;
    if (i < n) p[i] = 0;
}

__global__ void f32_to_bf16(const float* in, __hip_bfloat16* out, int n) {
    int i = blockIdx.x * blockDim.x + threadIdx.x;
    if (i < n) out[i] = __float2bfloat16(in[i]);
}

// W [K, Nc] row-major f32  ->  WT [Nc, K] row-major bf16  (WT[n*K+k] = W[k*Nc+n])
__global__ void transpose_to_bf16(const float* w, __hip_bfloat16* wt, int K, int Nc) {
    int idx = blockIdx.x * blockDim.x + threadIdx.x;
    if (idx >= K * Nc) return;
    int n = idx / K;
    int k = idx - n * K;
    wt[(size_t)n * K + k] = __float2bfloat16(w[(size_t)k * Nc + n]);
}

// ---------------------------------------------------------------------------
// Edge aggregation: agg[dst] += h[src]   (atomics land in L2: agg fits in 192MB)
// ---------------------------------------------------------------------------
__global__ void edge_agg_f32(const float* __restrict__ h,
                             const int* __restrict__ src, const int* __restrict__ dst,
                             float* __restrict__ agg, int H, long total) {
    long idx = (long)blockIdx.x * blockDim.x + threadIdx.x;
    if (idx >= total) return;
    int e = (int)(idx / H);
    int f = (int)(idx - (long)e * H);
    int s = src[e], d = dst[e];
    atomicAdd(&agg[(size_t)d * H + f], h[(size_t)s * H + f]);
}

__global__ void edge_agg_bf16(const __hip_bfloat16* __restrict__ h,
                              const int* __restrict__ src, const int* __restrict__ dst,
                              float* __restrict__ agg, int H, long total) {
    long idx = (long)blockIdx.x * blockDim.x + threadIdx.x;
    if (idx >= total) return;
    int e = (int)(idx / H);
    int f = (int)(idx - (long)e * H);
    int s = src[e], d = dst[e];
    atomicAdd(&agg[(size_t)d * H + f], __bfloat162float(h[(size_t)s * H + f]));
}

// ---------------------------------------------------------------------------
// Fused GraphConv GEMM:
//   out = tanh( A1[N,K1] @ W1 + A2[N,K2] @ W2 + bias ),  W given transposed bf16.
// One wave computes a 16x64 output strip. Per K-step we issue ALL fragment
// loads (2xA b128 + 8xB b128) before the 4 WMMAs so the backend can emit a
// single load clause + one s_wait, then 4 back-to-back v_wmma (XDL co-exec).
// Fragment layouts used here (wave32; A/B use contiguous K, not the ISA table order):
//   A: lane l holds row (l&15); halves 0..15 = contiguous K[k0 + (l>>4)*16 ..].
//   B: lane l holds col (l&15); halves 0..15 = contiguous K[k0 + (l>>4)*16 ..] of WT row.
//   D: lane l -> col (l&15); vgpr v -> row v + 8*(l>>4).
// ---------------------------------------------------------------------------
__global__ __launch_bounds__(256) void gemm_conv(
    const __hip_bfloat16* __restrict__ A1, const __hip_bfloat16* __restrict__ W1T, int K1,
    const __hip_bfloat16* __restrict__ A2, const __hip_bfloat16* __restrict__ W2T, int K2,
    const float* __restrict__ bias, __hip_bfloat16* __restrict__ out, int mblocks)
{
    const int H = HID;
    int wave = threadIdx.x >> 5;
    int lane = threadIdx.x & 31;
    int mblk = blockIdx.x % mblocks;        // 8 M-tiles per block
    int ngrp = blockIdx.x / mblocks;        // 64 output cols per block
    int m0 = (mblk * 8 + wave) * 16;
    int n0 = ngrp * 64;
    int rc = lane & 15;                     // row for A, col for B/D
    int hi = lane >> 4;

    v8f acc[4] = {};

    #pragma unroll
    for (int pass = 0; pass < 2; ++pass) {
        const __hip_bfloat16* A  = pass ? A2 : A1;
        const __hip_bfloat16* WT = pass ? W2T : W1T;
        const int K = pass ? K2 : K1;
        const char* ap = (const char*)(A + (size_t)(m0 + rc) * K + hi * 16);
        const char* bp = (const char*)(WT + (size_t)(n0 + rc) * K + hi * 16);
        const size_t bstr = (size_t)16 * K * 2;   // bytes between 16-col groups
        for (int k0 = 0; k0 < K; k0 += 32, ap += 64, bp += 64) {
            FragBF a, b0, b1, b2, b3;
            // ---- issue every load before any WMMA (one clause, one wait) ----
            a.u[0]  = ((const uint4*)ap)[0];
            a.u[1]  = ((const uint4*)ap)[1];
            b0.u[0] = ((const uint4*)bp)[0];
            b0.u[1] = ((const uint4*)bp)[1];
            b1.u[0] = ((const uint4*)(bp + bstr))[0];
            b1.u[1] = ((const uint4*)(bp + bstr))[1];
            b2.u[0] = ((const uint4*)(bp + 2 * bstr))[0];
            b2.u[1] = ((const uint4*)(bp + 2 * bstr))[1];
            b3.u[0] = ((const uint4*)(bp + 3 * bstr))[0];
            b3.u[1] = ((const uint4*)(bp + 3 * bstr))[1];
            // ---- 4 back-to-back WMMAs, disjoint accumulators ----
            acc[0] = __builtin_amdgcn_wmma_f32_16x16x32_bf16(
                false, a.v, false, b0.v, (short)0, acc[0], false, false);
            acc[1] = __builtin_amdgcn_wmma_f32_16x16x32_bf16(
                false, a.v, false, b1.v, (short)0, acc[1], false, false);
            acc[2] = __builtin_amdgcn_wmma_f32_16x16x32_bf16(
                false, a.v, false, b2.v, (short)0, acc[2], false, false);
            acc[3] = __builtin_amdgcn_wmma_f32_16x16x32_bf16(
                false, a.v, false, b3.v, (short)0, acc[3], false, false);
        }
    }

    #pragma unroll
    for (int t = 0; t < 4; ++t) {
        int c = n0 + t * 16 + rc;
        float bc = bias[c];
        #pragma unroll
        for (int v = 0; v < 8; ++v) {
            int r = m0 + v + 8 * hi;
            out[(size_t)r * H + c] = __float2bfloat16(tanhf(acc[t][v] + bc));
        }
    }
}

// ---------------------------------------------------------------------------
// Pooling (batch_index is sorted -> contiguous node ranges per graph)
// ---------------------------------------------------------------------------
__global__ void count_nodes(const int* __restrict__ batch, int* __restrict__ cnt, int n) {
    int i = blockIdx.x * blockDim.x + threadIdx.x;
    if (i < n) atomicAdd(&cnt[batch[i]], 1);
}

__global__ void scan_offsets(const int* __restrict__ cnt, int* __restrict__ off) {
    if (threadIdx.x == 0 && blockIdx.x == 0) {
        int acc = 0;
        off[0] = 0;
        for (int g = 0; g < N_GRAPH; ++g) { acc += cnt[g]; off[g + 1] = acc; }
    }
}

// gcat[g, 0:512] = segment_max, gcat[g, 512:1024] = segment_mean
__global__ void pool_kernel(const __hip_bfloat16* __restrict__ h,
                            const int* __restrict__ off, float* __restrict__ gcat) {
    int g = blockIdx.x >> 1;
    int f = ((blockIdx.x & 1) << 8) + threadIdx.x;   // 0..511
    int s = off[g], e = off[g + 1];
    float mx = -3.402823e38f, sum = 0.0f;
    for (int i = s; i < e; ++i) {
        float v = __bfloat162float(h[(size_t)i * HID + f]);
        mx = fmaxf(mx, v);
        sum += v;
    }
    int cnt = e - s;
    gcat[(size_t)g * (2 * HID) + f]       = (cnt > 0) ? mx : 0.0f;
    gcat[(size_t)g * (2 * HID) + HID + f] = sum / fmaxf((float)cnt, 1.0f);
}

// ---------------------------------------------------------------------------
// MLP tail (64 rows; negligible work -> plain f32 for exactness)
// ---------------------------------------------------------------------------
__global__ void mlp_lin(const float* __restrict__ in, const float* __restrict__ w,
                        const float* __restrict__ b, float* __restrict__ out,
                        int rows, int K, int Nc) {
    int idx = blockIdx.x * blockDim.x + threadIdx.x;
    if (idx >= rows * Nc) return;
    int r = idx / Nc, c = idx - (idx / Nc) * Nc;
    float s = b[c];
    for (int k = 0; k < K; ++k) s += in[(size_t)r * K + k] * w[(size_t)k * Nc + c];
    out[idx] = tanhf(s);
}

__global__ void lin3_logsoftmax(const float* __restrict__ in, const float* __restrict__ w,
                                const float* __restrict__ b, float* __restrict__ out) {
    __shared__ float sm[12];
    int g = blockIdx.x;
    int c = threadIdx.x;
    if (c < N_CLS) {
        float s = b[c];
        for (int k = 0; k < HID; ++k) s += in[(size_t)g * HID + k] * w[(size_t)k * N_CLS + c];
        sm[c] = s;
    }
    __syncthreads();
    if (c == 0) {
        float mx = -3.402823e38f;
        for (int i = 0; i < N_CLS; ++i) mx = fmaxf(mx, sm[i]);
        float sum = 0.0f;
        for (int i = 0; i < N_CLS; ++i) sum += expf(sm[i] - mx);
        sm[10] = mx + logf(sum);
    }
    __syncthreads();
    if (c < N_CLS) out[(size_t)g * N_CLS + c] = sm[c] - sm[10];
}

// ---------------------------------------------------------------------------
// Launch
// ---------------------------------------------------------------------------
extern "C" void kernel_launch(void* const* d_in, const int* in_sizes, int n_in,
                              void* d_out, int out_size, void* d_ws, size_t ws_size,
                              hipStream_t stream) {
    const float* x          = (const float*)d_in[0];
    const int*   edge_index = (const int*)d_in[1];
    const int*   batch      = (const int*)d_in[2];
    const float* w_root0    = (const float*)d_in[3];
    const float* w_rel0     = (const float*)d_in[4];
    const float* b0         = (const float*)d_in[5];
    const float* w_root     = (const float*)d_in[6];
    const float* w_rel      = (const float*)d_in[7];
    const float* bvec       = (const float*)d_in[8];
    const float* lin1_w     = (const float*)d_in[9];
    const float* lin1_b     = (const float*)d_in[10];
    const float* lin2_w     = (const float*)d_in[11];
    const float* lin2_b     = (const float*)d_in[12];
    const float* lin3_w     = (const float*)d_in[13];
    const float* lin3_b     = (const float*)d_in[14];
    float*       out        = (float*)d_out;

    const int N = N_NODES, E = N_EDGES, F = F_IN, H = HID, G = N_GRAPH;
    const int* src = edge_index;
    const int* dst = edge_index + E;

    // ---- workspace carve-out (256B aligned) ----
    size_t off = 0;
    auto wsalloc = [&](size_t bytes) -> char* {
        char* p = (char*)d_ws + off;
        off += (bytes + 255) & ~(size_t)255;
        return p;
    };
    __hip_bfloat16* x_bf     = (__hip_bfloat16*)wsalloc((size_t)N * F * 2);
    float*          agg0_f   = (float*)         wsalloc((size_t)N * F * 4);
    __hip_bfloat16* agg0_bf  = (__hip_bfloat16*)wsalloc((size_t)N * F * 2);
    __hip_bfloat16* hA       = (__hip_bfloat16*)wsalloc((size_t)N * H * 2);
    __hip_bfloat16* hB       = (__hip_bfloat16*)wsalloc((size_t)N * H * 2);
    float*          agg_f    = (float*)         wsalloc((size_t)N * H * 4);
    __hip_bfloat16* agg_bf   = (__hip_bfloat16*)wsalloc((size_t)N * H * 2);
    __hip_bfloat16* w0rootT  = (__hip_bfloat16*)wsalloc((size_t)F * H * 2);
    __hip_bfloat16* w0relT   = (__hip_bfloat16*)wsalloc((size_t)F * H * 2);
    __hip_bfloat16* wrootT   = (__hip_bfloat16*)wsalloc((size_t)4 * H * H * 2);
    __hip_bfloat16* wrelT    = (__hip_bfloat16*)wsalloc((size_t)4 * H * H * 2);
    float*          gcat     = (float*)         wsalloc((size_t)G * 2 * H * 4);
    float*          t1       = (float*)         wsalloc((size_t)G * H * 4);
    float*          t2       = (float*)         wsalloc((size_t)G * H * 4);
    int*            cnt      = (int*)           wsalloc((size_t)G * 4);
    int*            goff     = (int*)           wsalloc((size_t)(G + 1) * 4);

    const int TB = 256;
    auto cdiv = [](long a, long b) { return (int)((a + b - 1) / b); };

    // ---- weight prep: transpose to [Ncol][K] bf16 ----
    transpose_to_bf16<<<cdiv((long)F * H, TB), TB, 0, stream>>>(w_root0, w0rootT, F, H);
    transpose_to_bf16<<<cdiv((long)F * H, TB), TB, 0, stream>>>(w_rel0,  w0relT,  F, H);
    for (int i = 0; i < 4; ++i) {
        transpose_to_bf16<<<cdiv((long)H * H, TB), TB, 0, stream>>>(
            w_root + (size_t)i * H * H, wrootT + (size_t)i * H * H, H, H);
        transpose_to_bf16<<<cdiv((long)H * H, TB), TB, 0, stream>>>(
            w_rel  + (size_t)i * H * H, wrelT  + (size_t)i * H * H, H, H);
    }
    f32_to_bf16<<<cdiv((long)N * F, TB), TB, 0, stream>>>(x, x_bf, N * F);

    const int mblocks = N / 128;                     // 8 M-tiles (128 rows) per block
    const int gemm_grid = mblocks * (H / 64);        // x 64-col groups

    // ---- initial conv: F -> H ----
    zero_f32<<<cdiv((long)N * F, TB), TB, 0, stream>>>(agg0_f, N * F);
    edge_agg_f32<<<cdiv((long)E * F, TB), TB, 0, stream>>>(x, src, dst, agg0_f, F, (long)E * F);
    f32_to_bf16<<<cdiv((long)N * F, TB), TB, 0, stream>>>(agg0_f, agg0_bf, N * F);
    gemm_conv<<<gemm_grid, TB, 0, stream>>>(x_bf, w0rootT, F, agg0_bf, w0relT, F, b0, hA, mblocks);

    // ---- conv1..conv4: H -> H ----
    __hip_bfloat16* hcur = hA;
    __hip_bfloat16* hnxt = hB;
    for (int i = 0; i < 4; ++i) {
        zero_f32<<<cdiv((long)N * H, TB), TB, 0, stream>>>(agg_f, N * H);
        edge_agg_bf16<<<cdiv((long)E * H, TB), TB, 0, stream>>>(hcur, src, dst, agg_f, H, (long)E * H);
        f32_to_bf16<<<cdiv((long)N * H, TB), TB, 0, stream>>>(agg_f, agg_bf, N * H);
        gemm_conv<<<gemm_grid, TB, 0, stream>>>(hcur, wrootT + (size_t)i * H * H, H,
                                                agg_bf, wrelT + (size_t)i * H * H, H,
                                                bvec + (size_t)i * H, hnxt, mblocks);
        __hip_bfloat16* tmp = hcur; hcur = hnxt; hnxt = tmp;
    }

    // ---- pooling (sorted batch_index -> ranges) ----
    zero_i32<<<1, G, 0, stream>>>(cnt, G);
    count_nodes<<<cdiv(N, TB), TB, 0, stream>>>(batch, cnt, N);
    scan_offsets<<<1, 32, 0, stream>>>(cnt, goff);
    pool_kernel<<<G * 2, TB, 0, stream>>>(hcur, goff, gcat);

    // ---- MLP tail ----
    mlp_lin<<<cdiv((long)G * H, TB), TB, 0, stream>>>(gcat, lin1_w, lin1_b, t1, G, 2 * H, H);
    mlp_lin<<<cdiv((long)G * H, TB), TB, 0, stream>>>(t1,   lin2_w, lin2_b, t2, G, H, H);
    lin3_logsoftmax<<<G, 32, 0, stream>>>(t2, lin3_w, lin3_b, out);

    (void)n_in; (void)in_sizes; (void)out_size; (void)ws_size;
}
